// ODERNNDynamic_84851373900314
// MI455X (gfx1250) — matrix operand order repaired and checked by reading, not compiled, not run on hardware
//
#include <hip/hip_runtime.h>
#include <hip/hip_bf16.h>

#define BB   256
#define TT   1024
#define NDYN 20
#define DH   64
#define DZ   128
#define HID  64

typedef __attribute__((ext_vector_type(16))) _Float16 v16h;
typedef __attribute__((ext_vector_type(8)))  _Float16 v8h;
typedef __attribute__((ext_vector_type(8)))  float    v8f;

// ---------------------------------------------------------------------------
// Weight packing: K x N row-major f32  ->  per-WMMA-fragment f16 layout.
// Tile (k0,n0) is 32x16; tile order: n-tile major, k-tile minor (k contiguous
// for the inner GEMM loop). Within a tile each lane's 16 halves are contiguous
// so the GEMM does ONE 32-byte vector load per B fragment.
//   lane l, elem j (0..15):  K = k0 + ((l>>4)<<3) + (j&7) + ((j>>3)<<4),  N = n0 + (l&15)
// Rows >= Ksrc are zero (pads W_ih from 104 to 128).
// ---------------------------------------------------------------------------
__global__ void pack_wfrag(const float* __restrict__ W, _Float16* __restrict__ Wf,
                           int K, int N, int Ksrc) {
  int i = blockIdx.x * 256 + threadIdx.x;
  int total = (K >> 5) * (N >> 4) * 512;
  if (i >= total) return;
  int tile = i >> 9;
  int r    = i & 511;
  int lane = r >> 4;
  int j    = r & 15;
  int kt   = K >> 5;
  int n0   = (tile / kt) << 4;
  int k0   = (tile % kt) << 5;
  int kk   = k0 + ((lane >> 4) << 3) + (j & 7) + ((j >> 3) << 4);
  int n    = n0 + (lane & 15);
  float v  = (kk < Ksrc) ? W[kk * N + n] : 0.0f;
  Wf[i] = (_Float16)v;
}

// A fragment (16x32 f16) from LDS row-major [16][lda] halves.
// lane<16: row=lane,    K = k0 + {0..7, 16..23}
// lane>=16: row=lane-16, K = k0 + {8..15, 24..31}
__device__ __forceinline__ v16h load_A_lds(const _Float16* A, int lda, int k0) {
  int lane = threadIdx.x & 31;
  const _Float16* p = A + (lane & 15) * lda + k0 + ((lane >> 4) << 3);
  v8h lo = *reinterpret_cast<const v8h*>(p);
  v8h hi = *reinterpret_cast<const v8h*>(p + 16);
  v16h a;
#pragma unroll
  for (int i = 0; i < 8; ++i) { a[i] = lo[i]; a[i + 8] = hi[i]; }
  return a;
}

// C/D 16x16 f32 layout: lane<16 -> M=0..7 in regs 0..7, N=lane;
//                       lane>=16 -> M=8..15,           N=lane-16
__device__ __forceinline__ void store_C(float* C, int ldc, int n0, v8f acc) {
  int lane = threadIdx.x & 31;
  int n    = n0 + (lane & 15);
  int mb   = (lane >> 4) << 3;
#pragma unroll
  for (int r = 0; r < 8; ++r) C[(mb + r) * ldc + n] = acc[r];
}

// 16xN = (16xK f16 in LDS) @ (KxN fragment-packed f16 in global), f32 out to LDS.
// 8 waves split the N tiles.
__device__ __forceinline__ void gemm(const _Float16* A, int lda,
                                     const _Float16* __restrict__ Wf,
                                     float* C, int ldc, int K, int N) {
  int wave = threadIdx.x >> 5;
  int lane = threadIdx.x & 31;
  int kt   = K >> 5;
  for (int n0 = wave << 4; n0 < N; n0 += 8 << 4) {
    v8f acc = {};
    const _Float16* bp = Wf + (size_t)(n0 >> 4) * kt * 512 + lane * 16;
#pragma unroll
    for (int kk = 0; kk < 4; ++kk) {
      if (kk >= kt) break;
      v16h a = load_A_lds(A, lda, kk << 5);
      v16h b = *reinterpret_cast<const v16h*>(bp + kk * 512);
      acc = __builtin_amdgcn_wmma_f32_16x16x32_f16(false, a, false, b,
                                                   (short)0, acc, false, false);
    }
    store_C(C, ldc, n0, acc);
  }
}

// Branch-free fast transcendentals (v_exp_f32 + v_rcp_f32), correct limits:
//   x -> +inf: e=inf, rcp=0 -> tanh=+1 ; x -> -inf: e=0 -> tanh=-1.
__device__ __forceinline__ float tanh_fast(float x) {
  float e = __expf(2.0f * x);
  return 1.0f - 2.0f * __builtin_amdgcn_rcpf(e + 1.0f);
}
__device__ __forceinline__ float sigm_fast(float x) {
  return __builtin_amdgcn_rcpf(1.0f + __expf(-x));
}

union RkGru {
  struct { float kacc[16 * DZ]; float kbuf[16 * DZ]; } rk;      // 16 KB
  struct { float gi[16 * 3 * DZ]; float gh[16 * 3 * DZ]; } gru; // 48 KB
};

__global__ __launch_bounds__(256)
void odernn_main(const float* __restrict__ z0, const float* __restrict__ t_dyn,
                 const float* __restrict__ Y, const float* __restrict__ M,
                 const float* __restrict__ Hq, const int* __restrict__ lens,
                 const float* __restrict__ b0, const float* __restrict__ b1,
                 const float* __restrict__ b2, const float* __restrict__ b_ih,
                 const float* __restrict__ b_hh,
                 const _Float16* __restrict__ W0f, const _Float16* __restrict__ W1f,
                 const _Float16* __restrict__ W2f, const _Float16* __restrict__ Wihf,
                 const _Float16* __restrict__ Whhf,
                 float* __restrict__ out) {
  __shared__ RkGru u;                    // 48 KB
  __shared__ float    zf[16 * DZ];       //  8 KB  (f32 latent state / z_ode)
  __shared__ _Float16 Ah[16 * DZ];       //  4 KB  (f16 A staging, K padded to 128)
  __shared__ _Float16 Hh[16 * HID];      //  2 KB  (f16 MLP hidden)
  __shared__ float dtb[16], tprev[16], tkb[16];

  const int tid = threadIdx.x;
  const int r0  = blockIdx.x * 16;       // 16 batch rows per workgroup
  float* Zout = out + (size_t)BB * DZ;   // outputs: [z_final (B,DZ)] [Z (B,T,DZ)]

  for (int idx = tid; idx < 16 * DZ; idx += 256) {
    int row = idx >> 7, c = idx & 127;
    zf[idx] = z0[(r0 + row) * DZ + c];
  }
  if (tid < 16) tprev[tid] = t_dyn[(size_t)(r0 + tid) * TT];
  __syncthreads();

  for (int k = 0; k < TT; ++k) {
    if (tid < 16) {
      float tk = t_dyn[(size_t)(r0 + tid) * TT + k];
      tkb[tid] = tk;
      dtb[tid] = fmaxf(tk - tprev[tid], 0.0f);
      // Prefetch next step's inputs so HBM latency overlaps the RK4 chain
      // (lowers to global_prefetch_b8; fetches whole cachelines).
      if (k + 1 < TT) {
        size_t g = (size_t)(r0 + tid) * TT + (k + 1);
        __builtin_prefetch(&Y[g * NDYN], 0, 0);
        __builtin_prefetch(&M[g * NDYN], 0, 0);
        __builtin_prefetch(&Hq[g * DH], 0, 0);
        __builtin_prefetch(&t_dyn[g], 0, 0);
      }
    }
    __syncthreads();

    // ---------------- RK4 over dz/ds = ode_f(z)*dt ----------------
#pragma unroll 1
    for (int s = 0; s < 4; ++s) {
      const float a_s = (s == 0) ? 0.0f : (s == 3 ? 1.0f : 0.5f);
      const float w_s = (s == 0 || s == 3) ? 1.0f : 2.0f;

      for (int idx = tid; idx < 16 * DZ; idx += 256) {
        float zin = zf[idx];
        if (s > 0) zin += a_s * u.rk.kbuf[idx];
        Ah[idx] = (_Float16)zin;
      }
      __syncthreads();

      gemm(Ah, DZ, W0f, u.rk.kbuf, HID, DZ, HID);      // (16x128)@(128x64)
      __syncthreads();
      for (int idx = tid; idx < 16 * HID; idx += 256)
        Hh[idx] = (_Float16)tanh_fast(u.rk.kbuf[idx] + b0[idx & 63]);
      __syncthreads();

      gemm(Hh, HID, W1f, u.rk.kbuf, HID, HID, HID);    // (16x64)@(64x64)
      __syncthreads();
      for (int idx = tid; idx < 16 * HID; idx += 256)
        Hh[idx] = (_Float16)tanh_fast(u.rk.kbuf[idx] + b1[idx & 63]);
      __syncthreads();

      gemm(Hh, HID, W2f, u.rk.kbuf, DZ, HID, DZ);      // (16x64)@(64x128)
      __syncthreads();
      for (int idx = tid; idx < 16 * DZ; idx += 256) {
        int row = idx >> 7, c = idx & 127;
        float kv = (u.rk.kbuf[idx] + b2[c]) * dtb[row];
        u.rk.kbuf[idx] = kv;
        u.rk.kacc[idx] = (s == 0) ? kv : u.rk.kacc[idx] + w_s * kv;
      }
      __syncthreads();
    }
    // z_ode = z + (k1 + 2k2 + 2k3 + k4)/6   (state updated in place)
    for (int idx = tid; idx < 16 * DZ; idx += 256)
      zf[idx] += u.rk.kacc[idx] * (1.0f / 6.0f);
    __syncthreads();

    // ---------------- GRU cell ----------------
    // inp = [Y | M | H] padded to K=128, staged f16
    for (int idx = tid; idx < 16 * DZ; idx += 256) {
      int row = idx >> 7, c = idx & 127;
      size_t g = (size_t)(r0 + row) * TT + k;
      float v;
      if (c < NDYN)               v = Y[g * NDYN + c];
      else if (c < 2 * NDYN)      v = M[g * NDYN + (c - NDYN)];
      else if (c < 2 * NDYN + DH) v = Hq[g * DH + (c - 2 * NDYN)];
      else                        v = 0.0f;
      Ah[idx] = (_Float16)v;
    }
    __syncthreads();
    gemm(Ah, DZ, Wihf, u.gru.gi, 3 * DZ, DZ, 3 * DZ);  // (16x128)@(128x384)
    __syncthreads();

    for (int idx = tid; idx < 16 * DZ; idx += 256)     // stage z_ode as f16
      Ah[idx] = (_Float16)zf[idx];
    __syncthreads();
    gemm(Ah, DZ, Whhf, u.gru.gh, 3 * DZ, DZ, 3 * DZ);  // (16x128)@(128x384)
    __syncthreads();

    for (int idx = tid; idx < 16 * DZ; idx += 256) {
      int row = idx >> 7, c = idx & 127;
      const float* gi = u.gru.gi + row * 3 * DZ;
      const float* gh = u.gru.gh + row * 3 * DZ;
      float rr = sigm_fast(gi[c]          + b_ih[c]          + gh[c]          + b_hh[c]);
      float uu = sigm_fast(gi[DZ + c]     + b_ih[DZ + c]     + gh[DZ + c]     + b_hh[DZ + c]);
      float nn = tanh_fast(gi[2 * DZ + c] + b_ih[2 * DZ + c] +
                           rr * (gh[2 * DZ + c] + b_hh[2 * DZ + c]));
      float h  = zf[idx];                        // z_ode (f32)
      float zn = (1.0f - uu) * nn + uu * h;
      float zo = (k < lens[r0 + row]) ? zn : h;  // inactive rows keep z_ode
      zf[idx] = zo;
      Zout[((size_t)(r0 + row) * TT + k) * DZ + c] = zo;
    }
    __syncthreads();
    if (tid < 16) tprev[tid] = tkb[tid];
    __syncthreads();
  }

  for (int idx = tid; idx < 16 * DZ; idx += 256) {
    int row = idx >> 7, c = idx & 127;
    out[(size_t)(r0 + row) * DZ + c] = zf[idx];
  }
}

extern "C" void kernel_launch(void* const* d_in, const int* in_sizes, int n_in,
                              void* d_out, int out_size, void* d_ws, size_t ws_size,
                              hipStream_t stream) {
  const float* z0    = (const float*)d_in[0];
  const float* t_dyn = (const float*)d_in[1];
  const float* Y     = (const float*)d_in[2];
  const float* M     = (const float*)d_in[3];
  const float* Hq    = (const float*)d_in[4];
  const int*   lens  = (const int*)  d_in[5];
  const float* W0    = (const float*)d_in[6];
  const float* b0    = (const float*)d_in[7];
  const float* W1    = (const float*)d_in[8];
  const float* b1    = (const float*)d_in[9];
  const float* W2    = (const float*)d_in[10];
  const float* b2    = (const float*)d_in[11];
  const float* Wih   = (const float*)d_in[12];
  const float* b_ih  = (const float*)d_in[13];
  const float* Whh   = (const float*)d_in[14];
  const float* b_hh  = (const float*)d_in[15];

  _Float16* ws   = (_Float16*)d_ws;
  _Float16* W0f  = ws;            // 128x64  = 8192 halves
  _Float16* W1f  = ws + 8192;     // 64x64   = 4096
  _Float16* W2f  = ws + 12288;    // 64x128  = 8192
  _Float16* Wihf = ws + 20480;    // 128x384 = 49152 (K padded 104->128)
  _Float16* Whhf = ws + 69632;    // 128x384 = 49152            (~232 KB total)

  {
    int t0 = (128 / 32) * (64 / 16) * 512;
    pack_wfrag<<<(t0 + 255) / 256, 256, 0, stream>>>(W0, W0f, 128, 64, 128);
    int t1 = (64 / 32) * (64 / 16) * 512;
    pack_wfrag<<<(t1 + 255) / 256, 256, 0, stream>>>(W1, W1f, 64, 64, 64);
    int t2 = (64 / 32) * (128 / 16) * 512;
    pack_wfrag<<<(t2 + 255) / 256, 256, 0, stream>>>(W2, W2f, 64, 128, 64);
    int t3 = (128 / 32) * (384 / 16) * 512;
    pack_wfrag<<<(t3 + 255) / 256, 256, 0, stream>>>(Wih, Wihf, 128, 384, 104);
    pack_wfrag<<<(t3 + 255) / 256, 256, 0, stream>>>(Whh, Whhf, 128, 384, 128);
  }

  odernn_main<<<BB / 16, 256, 0, stream>>>(z0, t_dyn, Y, M, Hq, lens,
                                           b0, b1, b2, b_ih, b_hh,
                                           W0f, W1f, W2f, Wihf, Whhf,
                                           (float*)d_out);
}
